// SynthesisLayer_8907762172312
// MI455X (gfx1250) — matrix operand order repaired and checked by reading, not compiled, not run on hardware
//
#include <hip/hip_runtime.h>

#define BN   16
#define CIN  512
#define COUT 512
#define WD   512
#define RES  64
#define NPIX 4096            // RES*RES
#define RANK 10
#define NSTY ((CIN + COUT) * RANK)   // 10240
#define CO_R (COUT * RANK)           // 5120

#define TM 128
#define TN 128
#define TK 32
#define LDSPITCH 40          // padded row pitch in bf16 halves (80 bytes)

typedef __attribute__((ext_vector_type(16))) __bf16 v16bf;
typedef __attribute__((ext_vector_type(2)))  __bf16 v2bf;
typedef __attribute__((ext_vector_type(8)))  float  v8f;

struct Frag32 { uint4 lo, hi; };   // 32 bytes == one v16bf fragment

// f32 pair -> packed bf16 (RNE). Prefer the hw v_cvt_pk_bf16_f32.
__device__ __forceinline__ unsigned pack2bf(float lo, float hi) {
#if __has_builtin(__builtin_amdgcn_cvt_pk_bf16_f32)
    v2bf p = __builtin_amdgcn_cvt_pk_bf16_f32(lo, hi);
    return __builtin_bit_cast(unsigned, p);
#else
    v2bf p = { (__bf16)lo, (__bf16)hi };
    return __builtin_bit_cast(unsigned, p);
#endif
}
__device__ __forceinline__ unsigned short f2bf(float f) {
    __bf16 h = (__bf16)f;
    return __builtin_bit_cast(unsigned short, h);
}

// ---------------------------------------------------------------------------
// K1: styles[b][n] = bias[n] + (1/sqrt(WD)) * sum_k w[b][k] * aw[n][k]
// ---------------------------------------------------------------------------
__global__ __launch_bounds__(256) void styles_kernel(
    const float* __restrict__ w, const float* __restrict__ aw,
    const float* __restrict__ abias, float* __restrict__ styles) {
    int idx = blockIdx.x * 256 + threadIdx.x;          // 0 .. BN*NSTY-1
    int b = idx / NSTY;
    int n = idx - b * NSTY;
    const float4* wp = (const float4*)(w + (size_t)b * WD);
    const float4* ap = (const float4*)(aw + (size_t)n * WD);
    float s = 0.f;
#pragma unroll 4
    for (int k = 0; k < WD / 4; ++k) {
        float4 a = ap[k], ww = wp[k];
        s += a.x * ww.x + a.y * ww.y + a.z * ww.z + a.w * ww.w;
    }
    styles[idx] = s * 0.044194173824159216f + abias[n];   // 1/sqrt(512)
}

// ---------------------------------------------------------------------------
// K2: per (b,o): mod = left(o,:) @ right(:,i) / sqrt(RANK)
//     W = weight[o][i] * (mod+1);  W /= (||W||_2 + 1e-8);  -> bf16
// ---------------------------------------------------------------------------
__global__ __launch_bounds__(256) void modw_kernel(
    const float* __restrict__ styles, const float* __restrict__ weight,
    unsigned short* __restrict__ Wbf) {
    __shared__ float lrow[RANK];
    __shared__ float red[256];
    int b = blockIdx.x >> 9;            // /512
    int o = blockIdx.x & 511;
    int tid = threadIdx.x;
    const float* st = styles + (size_t)b * NSTY;
    if (tid < RANK) lrow[tid] = st[o * RANK + tid];
    __syncthreads();

    float vals[2];
    float ssq = 0.f;
#pragma unroll
    for (int j = 0; j < 2; ++j) {
        int i = tid + j * 256;
        float m = 0.f;
#pragma unroll
        for (int r = 0; r < RANK; ++r) m += lrow[r] * st[CO_R + r * CIN + i];
        m = m * 0.31622776601683794f + 1.0f;          // /sqrt(10), +1
        float Wv = weight[(size_t)o * CIN + i] * m;
        vals[j] = Wv;
        ssq += Wv * Wv;
    }
    red[tid] = ssq;
    __syncthreads();
#pragma unroll
    for (int s = 128; s > 0; s >>= 1) {
        if (tid < s) red[tid] += red[tid + s];
        __syncthreads();
    }
    float inv = 1.0f / (sqrtf(red[0]) + 1e-8f);
    unsigned short* dst = Wbf + ((size_t)b * COUT + o) * CIN;
#pragma unroll
    for (int j = 0; j < 2; ++j) dst[tid + j * 256] = f2bf(vals[j] * inv);
}

// ---------------------------------------------------------------------------
// K3: out[b][o][p] = lrelu( sum_i W[b][o][i]*x[b][i][p] + noise[p]*ns + bias[o] )*sqrt2
// bf16 WMMA, 128x128 block tile, 8 waves each computing 32x64 (2x4 MMAs).
// Software-pipelined: global loads for tile k+1 issued before computing tile k.
// ---------------------------------------------------------------------------
__global__ __launch_bounds__(256) void fmm_gemm_kernel(
    const float* __restrict__ x, const unsigned short* __restrict__ Wbf,
    const float* __restrict__ noise, const float* __restrict__ nstrength,
    const float* __restrict__ bias, float* __restrict__ out) {
    __shared__ __align__(16) unsigned short As[TM * LDSPITCH];   // [m][k]
    __shared__ __align__(16) unsigned short Bs[TN * LDSPITCH];   // [n][k] (transposed x)

    const int b   = blockIdx.z;
    const int o0  = blockIdx.y * TM;
    const int p0  = blockIdx.x * TN;
    const int tid = threadIdx.x;
    const int lane = tid & 31;
    const int wave = tid >> 5;           // 0..7
    const int wm = wave >> 1;            // 0..3 : row group (32 rows)
    const int wn = wave & 1;             // 0..1 : col group (64 cols)
    const int lhi  = lane >> 4;          // 0/1
    const int lrow = lane & 15;

    const unsigned short* Ag = Wbf + ((size_t)b * COUT + o0) * CIN;
    const float* xb = x + (size_t)b * CIN * NPIX;

    // staging-register mappings (fixed per thread)
    const int ar = tid >> 1;                // A row 0..127
    const int ac = (tid & 1) * 16;          // A col half-offset 0 or 16
    const int kp0 = tid >> 5;               // B row-pair for j=0 (0..7)
    const int kp1 = (tid + 256) >> 5;       // B row-pair for j=1 (8..15)
    const int c4  = tid & 31;               // B float4 column

    uint4  aReg[2];
    float4 bReg[2][2];                      // [j][row-in-pair]

    auto gload = [&](int kt) {
        const uint4* asrc = (const uint4*)(Ag + (size_t)ar * CIN + kt + ac);
        aReg[0] = asrc[0];
        aReg[1] = asrc[1];
        bReg[0][0] = *(const float4*)(xb + (size_t)(kt + 2 * kp0 + 0) * NPIX + p0 + c4 * 4);
        bReg[0][1] = *(const float4*)(xb + (size_t)(kt + 2 * kp0 + 1) * NPIX + p0 + c4 * 4);
        bReg[1][0] = *(const float4*)(xb + (size_t)(kt + 2 * kp1 + 0) * NPIX + p0 + c4 * 4);
        bReg[1][1] = *(const float4*)(xb + (size_t)(kt + 2 * kp1 + 1) * NPIX + p0 + c4 * 4);
    };

    v8f acc[2][4] = {};

    gload(0);                               // prologue prefetch

    for (int kt = 0; kt < CIN; kt += TK) {
        __syncthreads();                    // LDS free to overwrite
        // ---- drain staged registers into LDS
        {
            uint4* dst = (uint4*)&As[ar * LDSPITCH + ac];
            dst[0] = aReg[0];
            dst[1] = aReg[1];
        }
#pragma unroll
        for (int j = 0; j < 2; ++j) {
            int kp = j ? kp1 : kp0;
            float4 r0 = bReg[j][0], r1 = bReg[j][1];
            unsigned* q0 = (unsigned*)&Bs[(c4 * 4 + 0) * LDSPITCH + 2 * kp];
            unsigned* q1 = (unsigned*)&Bs[(c4 * 4 + 1) * LDSPITCH + 2 * kp];
            unsigned* q2 = (unsigned*)&Bs[(c4 * 4 + 2) * LDSPITCH + 2 * kp];
            unsigned* q3 = (unsigned*)&Bs[(c4 * 4 + 3) * LDSPITCH + 2 * kp];
            *q0 = pack2bf(r0.x, r1.x);      // halves (k, k+1) for pixel n
            *q1 = pack2bf(r0.y, r1.y);
            *q2 = pack2bf(r0.z, r1.z);
            *q3 = pack2bf(r0.w, r1.w);
        }
        // ---- prefetch next tile (in flight during WMMA phase)
        if (kt + TK < CIN) gload(kt + TK);
        __syncthreads();                    // LDS tile visible

        // ---- fragments + 8 WMMAs
        Frag32 afr[2], bfr[4];
#pragma unroll
        for (int mi = 0; mi < 2; ++mi) {
            int row = wm * 32 + mi * 16 + lrow;
            // 16-bit A 16x32 layout: lane half holds K runs [8*lhi..+7] and [16+8*lhi..+7]
            afr[mi].lo = *(const uint4*)&As[row * LDSPITCH + lhi * 8];
            afr[mi].hi = *(const uint4*)&As[row * LDSPITCH + 16 + lhi * 8];
        }
#pragma unroll
        for (int ni = 0; ni < 4; ++ni) {
            int col = wn * 64 + ni * 16 + lrow;
            // 16-bit B 32x16 layout: lane half holds K runs [8*lhi..+7] and [16+8*lhi..+7] of column N
            bfr[ni].lo = *(const uint4*)&Bs[col * LDSPITCH + lhi * 8];
            bfr[ni].hi = *(const uint4*)&Bs[col * LDSPITCH + 16 + lhi * 8];
        }
#pragma unroll
        for (int mi = 0; mi < 2; ++mi) {
            v16bf av = __builtin_bit_cast(v16bf, afr[mi]);
#pragma unroll
            for (int ni = 0; ni < 4; ++ni) {
                v16bf bv = __builtin_bit_cast(v16bf, bfr[ni]);
                acc[mi][ni] = __builtin_amdgcn_wmma_f32_16x16x32_bf16(
                    false, av, false, bv, (short)0, acc[mi][ni], false, false);
            }
        }
    }

    // ---- epilogue: + noise*ns, + bias, lrelu(0.2)*sqrt2 (non-temporal stores)
    const float ns = nstrength[0];
#pragma unroll
    for (int mi = 0; mi < 2; ++mi) {
#pragma unroll
        for (int ni = 0; ni < 4; ++ni) {
            int col = p0 + wn * 64 + ni * 16 + lrow;       // pixel index (h*RES+w)
            float nz = noise[col] * ns;
#pragma unroll
            for (int r = 0; r < 8; ++r) {
                int row = o0 + wm * 32 + mi * 16 + r + 8 * lhi;   // C/D layout: M=r+8*(lane>=16)
                float v = acc[mi][ni][r] + nz + bias[row];
                v = (v > 0.f ? v : 0.2f * v) * 1.4142135623730951f;
                __builtin_nontemporal_store(v, &out[((size_t)b * COUT + row) * NPIX + col]);
            }
        }
    }
}

// ---------------------------------------------------------------------------
extern "C" void kernel_launch(void* const* d_in, const int* in_sizes, int n_in,
                              void* d_out, int out_size, void* d_ws, size_t ws_size,
                              hipStream_t stream) {
    const float* x       = (const float*)d_in[0];   // [16,512,64,64]
    const float* w       = (const float*)d_in[1];   // [16,512]
    const float* aw      = (const float*)d_in[2];   // [10240,512]
    const float* abias   = (const float*)d_in[3];   // [10240]
    const float* weight  = (const float*)d_in[4];   // [512,512,1,1]
    const float* noise   = (const float*)d_in[5];   // [64,64]
    const float* nstr    = (const float*)d_in[6];   // scalar
    const float* bias    = (const float*)d_in[7];   // [512]
    float* out           = (float*)d_out;

    float*          styles = (float*)d_ws;                                  // 640 KB
    unsigned short* Wbf    = (unsigned short*)((char*)d_ws + (1u << 20));   // 8 MB

    styles_kernel<<<(BN * NSTY) / 256, 256, 0, stream>>>(w, aw, abias, styles);
    modw_kernel<<<BN * COUT, 256, 0, stream>>>(styles, weight, Wbf);

    dim3 grid(NPIX / TN, COUT / TM, BN);   // (32, 4, 16)
    fmm_gemm_kernel<<<grid, 256, 0, stream>>>(x, Wbf, noise, nstr, bias, out);
}
